// WorldModel_3762391351941
// MI455X (gfx1250) — matrix operand order repaired and checked by reading, not compiled, not run on hardware
//
#include <hip/hip_runtime.h>

typedef _Float16 h16;
typedef __attribute__((ext_vector_type(16))) _Float16 v16h;
typedef __attribute__((ext_vector_type(8)))  float    v8f;

#define Tn   512
#define Bn   128
#define OBSn 1024
#define ACTn 16
#define Hn   256
#define Sn   32
#define Fn   1442   // h(256) z(32) prm(32) prs(32) pm(32) ps(32) po(1024) pr(1) pc(1)

// f16 workspace offsets (in halves); weights stored transposed [N][Kpad]
#define OFF_RZ     0        // 512x320  : W_ih[:, :512] + pad(W_hh[:, :512])  (r|z fused)
#define OFF_IN     163840   // 256x320  : W_ih[:, 512:768]
#define OFF_HN     245760   // 256x256  : W_hh[:, 512:768]
#define OFF_POST1  311296   // 256x1280
#define OFF_POST2  638976   // 64x256
#define OFF_PRI1   655360   // 256x256
#define OFF_PRI2   720896   // 64x256
#define OFF_DOBS1  737280   // 256x288
#define OFF_DOBS2  811008   // 1024x256
#define OFF_DREW1  1073152  // 256x288
#define OFF_DREW2  1146880  // 16x256 (N padded 1->16)
#define OFF_DCONT1 1150976  // 256x288
#define OFF_DCONT2 1224704  // 16x256
#define WS_WEIGHT_BYTES 2457600          // 1228800 halves
#define OBSPRE_BYTES (512L*128*256*4)    // 64MB f32 partial sums (optional)

__device__ __forceinline__ v8f wmma16(v16h a, v16h b, v8f c) {
  return __builtin_amdgcn_wmma_f32_16x16x32_f16(false, a, false, b, (short)0, c, false, false);
}

// B fragment (32x16 f16, K x N): lane n = l&15 (+16 lanes repeat), dwords v0..3 = K[k0+8hi .. +7],
// v4..7 = K[k0+16+8hi .. +7]. Wt is [N][Kpad] -> 2x b128.
__device__ __forceinline__ v16h load_b(const h16* __restrict__ wt, int kpad, int n0, int k0, int lane) {
  const h16* p = wt + (long)(n0 + (lane & 15)) * kpad + k0 + 8 * (lane >> 4);
  union { v16h v; uint4 q[2]; } f;
  f.q[0] = *(const uint4*)(p);
  f.q[1] = *(const uint4*)(p + 16);
  return f.v;
}

// A fragment (16x32 f16, M x K) from LDS row-major tile: lane row = l&15,
// dwords v0..3 = K[k0+8hi .. +7], v4..7 = K[k0+16+8hi .. +7]  -> 2x ds_load_b128
__device__ __forceinline__ v16h load_a_lds(const h16* p0, int stride, int r0, int k0, int lane) {
  const h16* p = p0 + (r0 + (lane & 15)) * stride + k0 + 8 * (lane >> 4);
  union { v16h v; uint4 q[2]; } f;
  f.q[0] = *(const uint4*)(p);
  f.q[1] = *(const uint4*)(p + 16);
  return f.v;
}

// A fragment from global f32, aligned fast path (obs): float4 loads + cvt to f16
__device__ __forceinline__ v16h load_a_gv(const float* base, long rs, int r0, int c0, int lane) {
  const float* rp = base + (long)(r0 + (lane & 15)) * rs + c0 + 8 * (lane >> 4);
  const float4 f0 = *(const float4*)(rp);
  const float4 f1 = *(const float4*)(rp + 4);
  const float4 f2 = *(const float4*)(rp + 16);
  const float4 f3 = *(const float4*)(rp + 20);
  union { v16h v; h16 h[16]; } f;
  f.h[0]=(h16)f0.x; f.h[1]=(h16)f0.y; f.h[2]=(h16)f0.z; f.h[3]=(h16)f0.w;
  f.h[4]=(h16)f1.x; f.h[5]=(h16)f1.y; f.h[6]=(h16)f1.z; f.h[7]=(h16)f1.w;
  f.h[8]=(h16)f2.x; f.h[9]=(h16)f2.y; f.h[10]=(h16)f2.z; f.h[11]=(h16)f2.w;
  f.h[12]=(h16)f3.x; f.h[13]=(h16)f3.y; f.h[14]=(h16)f3.z; f.h[15]=(h16)f3.w;
  return f.v;
}

// A fragment from global f32, per-element with column limit (cols >= ncols -> 0)
__device__ __forceinline__ v16h load_a_ge(const float* base, long rs, int r0, int c0, int ncols, int lane) {
  const float* rp = base + (long)(r0 + (lane & 15)) * rs;
  const int o1 = c0 + 8 * (lane >> 4);
  const int o2 = o1 + 16;
  union { v16h v; h16 h[16]; } f;
#pragma unroll
  for (int i = 0; i < 8; ++i) {
    const int c1 = o1 + i, c2 = o2 + i;
    f.h[i]     = (c1 < ncols) ? (h16)rp[c1] : (h16)0.0f;
    f.h[8 + i] = (c2 < ncols) ? (h16)rp[c2] : (h16)0.0f;
  }
  return f.v;
}

__device__ __forceinline__ float sigmf(float x)  { return 1.0f / (1.0f + __expf(-x)); }
__device__ __forceinline__ float softpf(float x) { return (x > 20.0f) ? x : log1pf(__expf(x)); }

// ---------------- weight prep: transpose + f32->f16, pad K (and N) ----------------
__global__ __launch_bounds__(256) void prep_transpose(
    const float* __restrict__ src, h16* __restrict__ dst,
    int K, int Nsrc, int Npad, int Kpad)
{
  const long i = (long)blockIdx.x * 256 + threadIdx.x;
  if (i >= (long)Npad * Kpad) return;
  const int nI = (int)(i / Kpad);
  const int k  = (int)(i % Kpad);
  float v = 0.0f;
  if (k < K && nI < Nsrc) v = src[(long)k * Nsrc + nI];
  dst[i] = (h16)v;
}

// GRU: fuse W_hh r/z columns into W_ih (x = [h,z,a] contains h), keep n-gate split
__global__ __launch_bounds__(256) void prep_gru(
    const float* __restrict__ W_ih, const float* __restrict__ W_hh, h16* __restrict__ ws)
{
  const long i = (long)blockIdx.x * 256 + threadIdx.x;
  const long NRZ = 512L * 320, NIN = 256L * 320, NHN = 256L * 256;
  if (i < NRZ) {
    const int nI = (int)(i / 320), k = (int)(i % 320);
    float v = 0.0f;
    if (k < 304) v = W_ih[(long)k * 768 + nI];
    if (k < 256) v += W_hh[(long)k * 768 + nI];
    ws[OFF_RZ + i] = (h16)v;
  } else if (i < NRZ + NIN) {
    const long j = i - NRZ;
    const int nI = (int)(j / 320), k = (int)(j % 320);
    ws[OFF_IN + j] = (h16)((k < 304) ? W_ih[(long)k * 768 + 512 + nI] : 0.0f);
  } else if (i < NRZ + NIN + NHN) {
    const long j = i - NRZ - NIN;
    const int nI = (int)(j / 256), k = (int)(j % 256);
    ws[OFF_HN + j] = (h16)W_hh[(long)k * 768 + 512 + nI];
  }
}

// ---------------- hoisted obs-side of posterior layer 1 (state-independent) ----------
// obs_pre[t][b][0:256] = obs[b,t,:] @ post_W1[256:1280,:]   (fully parallel over t,b)
__global__ __launch_bounds__(256) void prep_post1_obs(
    const float* __restrict__ obs, const h16* __restrict__ ws, float* __restrict__ obs_pre)
{
  const int t  = blockIdx.x >> 3;
  const int mt = blockIdx.x & 7;
  const int r0 = mt * 16;
  const int wave = threadIdx.x >> 5, lane = threadIdx.x & 31;
  const int n = lane & 15, hi = lane >> 4;
  const float* obase = obs + (long)t * OBSn;
  for (int ct = wave; ct < 16; ct += 8) {
    const int n0 = ct * 16;
    v8f acc = {};
    for (int kt = 0; kt < 32; ++kt)
      acc = wmma16(load_a_gv(obase, (long)Tn * OBSn, r0, kt * 32, lane),
                   load_b(ws + OFF_POST1, 1280, n0, 256 + kt * 32, lane), acc);
#pragma unroll
    for (int v = 0; v < 8; ++v)
      obs_pre[((long)t * Bn + r0 + v + 8 * hi) * Hn + n0 + n] = acc[v];
  }
}

// ---------------- sequential recurrence: one persistent workgroup, LDS-resident state ----
__global__ __launch_bounds__(512) void rssm_recurrence(
    const float* __restrict__ obs, const float* __restrict__ act,
    const float* __restrict__ eps,
    const float* __restrict__ b_ih, const float* __restrict__ b_hh,
    const float* __restrict__ post_b1, const float* __restrict__ post_b2,
    const h16* __restrict__ ws, const float* __restrict__ obs_pre,
    float* __restrict__ out)
{
  __shared__ h16 s_h[2][Bn * Hn];   // 2 x 64KB, ping-pong hidden state (f16)
  __shared__ h16 s_z[Bn * Sn];      // 8KB stochastic state
  __shared__ h16 s_hid[Bn * Hn];    // 64KB posterior hidden

  const int tid  = threadIdx.x;
  const int wave = tid >> 5;
  const int lane = tid & 31;
  const int n    = lane & 15;
  const int hi   = lane >> 4;

  for (int i = tid; i < Bn * Hn; i += 512) s_h[0][i] = (h16)0.0f;
  __syncthreads();

  for (int t = 0; t < Tn; ++t) {
    const int cur = t & 1, prev = cur ^ 1;

    if (t == 0) {
      // h_0 = 0 (GRU skipped when is_first)
      for (int i = tid; i < Bn * Hn; i += 512) {
        const int b = i >> 8, c = i & 255;
        out[((long)b * Tn) * Fn + c] = 0.0f;
      }
    } else {
      const float* abase = act + (long)(t - 1) * ACTn;
      // 8 row-tiles x 16 c-tiles; each group keeps r,z,i_n,h_n accumulators in VGPRs
      for (int g = wave; g < 128; g += 16) {
        const int mt = g & 7, ct = g >> 3;
        const int r0 = mt * 16;
        v8f ar = {}, az = {}, ai = {}, ah = {};
#pragma unroll
        for (int kt = 0; kt < 10; ++kt) {       // x = [h(256) | z(32) | a(16) pad(16)]
          const int k0 = kt * 32;
          v16h a;
          if (kt < 8)       a = load_a_lds(s_h[prev], Hn, r0, k0, lane);
          else if (kt == 8) a = load_a_lds(s_z, Sn, r0, 0, lane);
          else              a = load_a_ge(abase, (long)Tn * ACTn, r0, 0, ACTn, lane);
          ar = wmma16(a, load_b(ws + OFF_RZ, 320, ct * 16,       k0, lane), ar);
          az = wmma16(a, load_b(ws + OFF_RZ, 320, 256 + ct * 16, k0, lane), az);
          ai = wmma16(a, load_b(ws + OFF_IN, 320, ct * 16,       k0, lane), ai);
        }
#pragma unroll
        for (int kt = 0; kt < 8; ++kt)
          ah = wmma16(load_a_lds(s_h[prev], Hn, r0, kt * 32, lane),
                      load_b(ws + OFF_HN, 256, ct * 16, kt * 32, lane), ah);
        const int c  = ct * 16 + n;
        const float br = b_ih[c] + b_hh[c];
        const float bz = b_ih[Hn + c] + b_hh[Hn + c];
        const float bi = b_ih[2 * Hn + c];
        const float bh = b_hh[2 * Hn + c];
#pragma unroll
        for (int v = 0; v < 8; ++v) {
          const int brow = r0 + v + 8 * hi;
          const float rg = sigmf(ar[v] + br);
          const float zg = sigmf(az[v] + bz);
          const float ng = tanhf(ai[v] + bi + rg * (ah[v] + bh));
          const float ho = (float)s_h[prev][brow * Hn + c];
          const float hn2 = (1.0f - zg) * ng + zg * ho;
          s_h[cur][brow * Hn + c] = (h16)hn2;
          out[((long)brow * Tn + t) * Fn + c] = hn2;
        }
      }
    }
    __syncthreads();

    // posterior layer 1: [h | obs] (K=1280) -> 256, ReLU, into LDS.
    // obs-side partial sums come precomputed (obs_pre) when scratch allows;
    // otherwise fall back to the in-loop obs GEMM.
    {
      const float* obase = obs + (long)t * OBSn;
      const float* pre = obs_pre ? (obs_pre + (long)t * Bn * Hn) : nullptr;
      for (int g = wave; g < 128; g += 16) {
        const int mt = g & 7, ct = g >> 3;
        const int r0 = mt * 16, n0 = ct * 16;
        v8f acc = {};
        if (pre) {
#pragma unroll
          for (int v = 0; v < 8; ++v)
            acc[v] = pre[(r0 + v + 8 * hi) * Hn + n0 + n];
        }
#pragma unroll
        for (int kt = 0; kt < 8; ++kt)
          acc = wmma16(load_a_lds(s_h[cur], Hn, r0, kt * 32, lane),
                       load_b(ws + OFF_POST1, 1280, n0, kt * 32, lane), acc);
        if (!pre) {
          for (int kt = 0; kt < 32; ++kt)
            acc = wmma16(load_a_gv(obase, (long)Tn * OBSn, r0, kt * 32, lane),
                         load_b(ws + OFF_POST1, 1280, n0, 256 + kt * 32, lane), acc);
        }
        const float bb = post_b1[n0 + n];
#pragma unroll
        for (int v = 0; v < 8; ++v) {
          const float x = acc[v] + bb;
          s_hid[(r0 + v + 8 * hi) * Hn + n0 + n] = (h16)(x > 0.0f ? x : 0.0f);
        }
      }
    }
    __syncthreads();

    // posterior layer 2 (256->64) + reparameterized sample; each wave owns a full row
    // tile so mean col j and std col j+32 land in the SAME lane (no exchange needed)
    if (wave < 8) {
      const int r0 = wave * 16;
      v8f a0 = {}, a1 = {}, a2 = {}, a3 = {};
#pragma unroll
      for (int kt = 0; kt < 8; ++kt) {
        const v16h a = load_a_lds(s_hid, Hn, r0, kt * 32, lane);
        a0 = wmma16(a, load_b(ws + OFF_POST2, 256, 0,  kt * 32, lane), a0);
        a1 = wmma16(a, load_b(ws + OFF_POST2, 256, 16, kt * 32, lane), a1);
        a2 = wmma16(a, load_b(ws + OFF_POST2, 256, 32, kt * 32, lane), a2);
        a3 = wmma16(a, load_b(ws + OFF_POST2, 256, 48, kt * 32, lane), a3);
      }
      const float bm0 = post_b2[n],      bm1 = post_b2[16 + n];
      const float bs0 = post_b2[32 + n], bs1 = post_b2[48 + n];
#pragma unroll
      for (int v = 0; v < 8; ++v) {
        const int brow = r0 + v + 8 * hi;
        const long ob = ((long)brow * Tn + t) * Fn;
        const float* ep = eps + ((long)t * Bn + brow) * Sn;
        const float pm0 = a0[v] + bm0, pm1 = a1[v] + bm1;
        const float ps0 = softpf(a2[v] + bs0) + 0.1f;
        const float ps1 = softpf(a3[v] + bs1) + 0.1f;
        const float z0 = pm0 + ps0 * ep[n];
        const float z1 = pm1 + ps1 * ep[16 + n];
        out[ob + 352 + n] = pm0;  out[ob + 368 + n] = pm1;
        out[ob + 384 + n] = ps0;  out[ob + 400 + n] = ps1;
        out[ob + 256 + n] = z0;   out[ob + 272 + n] = z1;
        s_z[brow * Sn + n]      = (h16)z0;
        s_z[brow * Sn + 16 + n] = (h16)z1;
      }
    }
    __syncthreads();
  }
}

// ---------------- parallel heads: prior + 3 decoders over 4096 (t, row-tile) blocks ----
__global__ __launch_bounds__(256) void rssm_heads(
    const float* __restrict__ prior_b1, const float* __restrict__ prior_b2,
    const float* __restrict__ dobs_b1,  const float* __restrict__ dobs_b2,
    const float* __restrict__ drew_b1,  const float* __restrict__ drew_b2,
    const float* __restrict__ dcont_b1, const float* __restrict__ dcont_b2,
    const h16* __restrict__ ws, float* __restrict__ out)
{
  __shared__ h16 s_feat[16 * 288];   // [h | z] tile, contiguous at out offsets 0..287
  __shared__ h16 s_hp[16 * 256];
  __shared__ h16 s_ho[16 * 256];
  __shared__ h16 s_hr[16 * 256];
  __shared__ h16 s_hc[16 * 256];

  const int t   = blockIdx.x >> 3;
  const int mt  = blockIdx.x & 7;
  const int r0g = mt * 16;
  const int tid  = threadIdx.x;
  const int wave = tid >> 5, lane = tid & 31;
  const int n = lane & 15, hi = lane >> 4;

  for (int i = tid; i < 16 * 288; i += 256) {
    const int r = i / 288, c = i % 288;
    s_feat[i] = (h16)out[((long)(r0g + r) * Tn + t) * Fn + c];
  }
  __syncthreads();

  // hidden layers: 4 heads x 16 c-tiles = 64 jobs over 8 waves
  for (int job = wave; job < 64; job += 8) {
    const int which = job >> 4;
    const int n0 = (job & 15) * 16;
    const h16* W; const float* bias; h16* dst; int kpad, ktot;
    if (which == 0)      { W = ws + OFF_PRI1;   kpad = 256; ktot = 8; bias = prior_b1; dst = s_hp; }
    else if (which == 1) { W = ws + OFF_DOBS1;  kpad = 288; ktot = 9; bias = dobs_b1;  dst = s_ho; }
    else if (which == 2) { W = ws + OFF_DREW1;  kpad = 288; ktot = 9; bias = drew_b1;  dst = s_hr; }
    else                 { W = ws + OFF_DCONT1; kpad = 288; ktot = 9; bias = dcont_b1; dst = s_hc; }
    v8f acc = {};
    for (int kt = 0; kt < ktot; ++kt)
      acc = wmma16(load_a_lds(s_feat, 288, 0, kt * 32, lane),
                   load_b(W, kpad, n0, kt * 32, lane), acc);
    const float bb = bias[n0 + n];
#pragma unroll
    for (int v = 0; v < 8; ++v) {
      const float x = acc[v] + bb;
      dst[(v + 8 * hi) * 256 + n0 + n] = (h16)(x > 0.0f ? x : 0.0f);
    }
  }
  __syncthreads();

  // output layers: 64 obs-decoder tiles + prior(4 accs) + reward + cont = 67 jobs
  for (int job = wave; job < 67; job += 8) {
    if (job < 64) {
      const int n0 = job * 16;
      v8f acc = {};
#pragma unroll
      for (int kt = 0; kt < 8; ++kt)
        acc = wmma16(load_a_lds(s_ho, 256, 0, kt * 32, lane),
                     load_b(ws + OFF_DOBS2, 256, n0, kt * 32, lane), acc);
      const float bb = dobs_b2[n0 + n];
#pragma unroll
      for (int v = 0; v < 8; ++v)
        out[((long)(r0g + v + 8 * hi) * Tn + t) * Fn + 416 + n0 + n] = acc[v] + bb;
    } else if (job == 64) {
      v8f a0 = {}, a1 = {}, a2 = {}, a3 = {};
#pragma unroll
      for (int kt = 0; kt < 8; ++kt) {
        const v16h a = load_a_lds(s_hp, 256, 0, kt * 32, lane);
        a0 = wmma16(a, load_b(ws + OFF_PRI2, 256, 0,  kt * 32, lane), a0);
        a1 = wmma16(a, load_b(ws + OFF_PRI2, 256, 16, kt * 32, lane), a1);
        a2 = wmma16(a, load_b(ws + OFF_PRI2, 256, 32, kt * 32, lane), a2);
        a3 = wmma16(a, load_b(ws + OFF_PRI2, 256, 48, kt * 32, lane), a3);
      }
      const float bm0 = prior_b2[n],      bm1 = prior_b2[16 + n];
      const float bs0 = prior_b2[32 + n], bs1 = prior_b2[48 + n];
#pragma unroll
      for (int v = 0; v < 8; ++v) {
        const long ob = ((long)(r0g + v + 8 * hi) * Tn + t) * Fn;
        out[ob + 288 + n] = a0[v] + bm0;
        out[ob + 304 + n] = a1[v] + bm1;
        out[ob + 320 + n] = softpf(a2[v] + bs0) + 0.1f;
        out[ob + 336 + n] = softpf(a3[v] + bs1) + 0.1f;
      }
    } else {
      const bool is_cont = (job == 66);
      const h16* hsrc = is_cont ? s_hc : s_hr;
      const h16* W = ws + (is_cont ? OFF_DCONT2 : OFF_DREW2);
      v8f acc = {};
#pragma unroll
      for (int kt = 0; kt < 8; ++kt)
        acc = wmma16(load_a_lds(hsrc, 256, 0, kt * 32, lane),
                     load_b(W, 256, 0, kt * 32, lane), acc);
      if (n == 0) {   // only column 0 is the real (unpadded) output
        const float b0 = is_cont ? dcont_b2[0] : drew_b2[0];
#pragma unroll
        for (int v = 0; v < 8; ++v) {
          float x = acc[v] + b0;
          if (is_cont) x = sigmf(x);
          out[((long)(r0g + v + 8 * hi) * Tn + t) * Fn + (is_cont ? 1441 : 1440)] = x;
        }
      }
    }
  }
}

extern "C" void kernel_launch(void* const* d_in, const int* in_sizes, int n_in,
                              void* d_out, int out_size, void* d_ws, size_t ws_size,
                              hipStream_t stream)
{
  (void)in_sizes; (void)n_in; (void)out_size;
  const float* obs      = (const float*)d_in[0];
  const float* act      = (const float*)d_in[1];
  const float* eps      = (const float*)d_in[2];
  const float* prior_W1 = (const float*)d_in[3];
  const float* prior_b1 = (const float*)d_in[4];
  const float* prior_W2 = (const float*)d_in[5];
  const float* prior_b2 = (const float*)d_in[6];
  const float* post_W1  = (const float*)d_in[7];
  const float* post_b1  = (const float*)d_in[8];
  const float* post_W2  = (const float*)d_in[9];
  const float* post_b2  = (const float*)d_in[10];
  const float* W_ih     = (const float*)d_in[11];
  const float* b_ih     = (const float*)d_in[12];
  const float* W_hh     = (const float*)d_in[13];
  const float* b_hh     = (const float*)d_in[14];
  const float* dobs_W1  = (const float*)d_in[15];
  const float* dobs_b1  = (const float*)d_in[16];
  const float* dobs_W2  = (const float*)d_in[17];
  const float* dobs_b2  = (const float*)d_in[18];
  const float* drew_W1  = (const float*)d_in[19];
  const float* drew_b1  = (const float*)d_in[20];
  const float* drew_W2  = (const float*)d_in[21];
  const float* drew_b2  = (const float*)d_in[22];
  const float* dcont_W1 = (const float*)d_in[23];
  const float* dcont_b1 = (const float*)d_in[24];
  const float* dcont_W2 = (const float*)d_in[25];
  const float* dcont_b2 = (const float*)d_in[26];
  h16* ws     = (h16*)d_ws;
  float* outp = (float*)d_out;

  // deterministic scratch-size gate for the hoisted obs GEMM (same every call)
  float* obs_pre = nullptr;
  if (ws_size >= (size_t)WS_WEIGHT_BYTES + (size_t)OBSPRE_BYTES)
    obs_pre = (float*)((char*)d_ws + WS_WEIGHT_BYTES);

  auto nb = [](long tot) { return (unsigned)((tot + 255) / 256); };

  prep_gru<<<nb(311296), 256, 0, stream>>>(W_ih, W_hh, ws);
  prep_transpose<<<nb(327680), 256, 0, stream>>>(post_W1,  ws + OFF_POST1, 1280, 256, 256, 1280);
  prep_transpose<<<nb(16384),  256, 0, stream>>>(post_W2,  ws + OFF_POST2, 256, 64, 64, 256);
  prep_transpose<<<nb(65536),  256, 0, stream>>>(prior_W1, ws + OFF_PRI1,  256, 256, 256, 256);
  prep_transpose<<<nb(16384),  256, 0, stream>>>(prior_W2, ws + OFF_PRI2,  256, 64, 64, 256);
  prep_transpose<<<nb(73728),  256, 0, stream>>>(dobs_W1,  ws + OFF_DOBS1, 288, 256, 256, 288);
  prep_transpose<<<nb(262144), 256, 0, stream>>>(dobs_W2,  ws + OFF_DOBS2, 256, 1024, 1024, 256);
  prep_transpose<<<nb(73728),  256, 0, stream>>>(drew_W1,  ws + OFF_DREW1, 288, 256, 256, 288);
  prep_transpose<<<nb(4096),   256, 0, stream>>>(drew_W2,  ws + OFF_DREW2, 256, 1, 16, 256);
  prep_transpose<<<nb(73728),  256, 0, stream>>>(dcont_W1, ws + OFF_DCONT1, 288, 256, 256, 288);
  prep_transpose<<<nb(4096),   256, 0, stream>>>(dcont_W2, ws + OFF_DCONT2, 256, 1, 16, 256);

  if (obs_pre)
    prep_post1_obs<<<Tn * 8, 256, 0, stream>>>(obs, ws, obs_pre);

  rssm_recurrence<<<1, 512, 0, stream>>>(obs, act, eps, b_ih, b_hh,
                                         post_b1, post_b2, ws, obs_pre, outp);
  rssm_heads<<<Tn * 8, 256, 0, stream>>>(prior_b1, prior_b2, dobs_b1, dobs_b2,
                                         drew_b1, drew_b2, dcont_b1, dcont_b2, ws, outp);
}
